// MaskedSelfAttention_23570780521145
// MI455X (gfx1250) — compile-verified, matrix operand order checked by reading, not run on hardware
//
#include <hip/hip_runtime.h>

#define B_N 4
#define S_N 2048
#define D_N 512
#define BS_N (B_N * S_N)

typedef __attribute__((ext_vector_type(16))) __bf16 v16bf;
typedef __attribute__((ext_vector_type(8)))  float  v8f;

union FragBF { v16bf v; unsigned int u[8]; };

// scheduler pipelining hints (mask: 0x008 = MFMA/WMMA, 0x020 = VMEM read)
#if defined(__has_builtin)
#if __has_builtin(__builtin_amdgcn_sched_group_barrier)
#define SCHED_GROUP(mask, size, id) __builtin_amdgcn_sched_group_barrier(mask, size, id)
#endif
#endif
#ifndef SCHED_GROUP
#define SCHED_GROUP(mask, size, id)
#endif

__device__ __forceinline__ unsigned short f2bf(float f) {
    union { float f; unsigned int u; } x; x.f = f;
    unsigned int u = x.u;
    u += 0x7fffu + ((u >> 16) & 1u);   // round to nearest even
    return (unsigned short)(u >> 16);
}

// Load a 16x32 bf16 A/B fragment. `p` is this lane's row base pointer (row-major,
// consecutive K elements contiguous). K pairs (2i,2i+1) sit in one dword.
// Layout per ISA 7.12.2: lanes 0-15 hold K 0..7 / 16..23, lanes 16-31 hold K 8..15 / 24..31.
__device__ __forceinline__ v16bf load_frag_pairs(const unsigned short* p, int half) {
    FragBF f;
#pragma unroll
    for (int i = 0; i < 8; ++i) {
        int base = (i < 4) ? (2 * i) : (16 + 2 * (i - 4));
        int k = base + half * 8;
        f.u[i] = *reinterpret_cast<const unsigned int*>(p + k);
    }
    return f.v;
}

__device__ __forceinline__ v8f wmma_bf16(v16bf a, v16bf b, v8f c) {
    return __builtin_amdgcn_wmma_f32_16x16x32_bf16(false, a, false, b, (short)0, c, false, false);
}

// ------------------------------------------------- fp32 -> bf16 (4 elems/thread)
__global__ void cvt_kernel(const float* __restrict__ src, unsigned short* __restrict__ dst, int n4) {
    int i = blockIdx.x * blockDim.x + threadIdx.x;
    if (i < n4) {
        float4 f = reinterpret_cast<const float4*>(src)[i];
        union { unsigned short h[4]; uint2 u; } o;
        o.h[0] = f2bf(f.x); o.h[1] = f2bf(f.y);
        o.h[2] = f2bf(f.z); o.h[3] = f2bf(f.w);
        reinterpret_cast<uint2*>(dst)[i] = o.u;
    }
}

// ------------------------------------------------------- fused QKV projection
// grid (BS/32, 3), block 256 (8 waves). Wave w computes a 32x64 output tile:
// rows [m0,m0+32), cols [64w,64w+64). Two A fragments share every B fragment.
// y = x @ W^T + b.  which==2 (V) stores transposed vt[b][d][s].
__global__ __launch_bounds__(256) __attribute__((amdgpu_waves_per_eu(1)))
void qkv_kernel(
        const unsigned short* __restrict__ xb,
        const unsigned short* __restrict__ wq, const unsigned short* __restrict__ wk,
        const unsigned short* __restrict__ wv,
        const float* __restrict__ bq, const float* __restrict__ bk, const float* __restrict__ bv,
        unsigned short* __restrict__ qo, unsigned short* __restrict__ ko,
        unsigned short* __restrict__ vto) {
    const int lane = threadIdx.x & 31;
    const int wave = threadIdx.x >> 5;
    const int idx = lane & 15, half = lane >> 4;
    const int m0 = blockIdx.x * 32;
    const int which = blockIdx.y;
    const unsigned short* W = (which == 0) ? wq : (which == 1) ? wk : wv;
    const float* bias = (which == 0) ? bq : (which == 1) ? bk : bv;
    const int n0 = wave * 64;

    v8f c[2][4] = {};
    const unsigned short* arow0 = xb + (size_t)(m0 + idx) * D_N;
    const unsigned short* arow1 = arow0 + (size_t)16 * D_N;
    const unsigned short* brow  = W + (size_t)(n0 + idx) * D_N;

#pragma unroll
    for (int k0 = 0; k0 < D_N; k0 += 32) {
        v16bf a0 = load_frag_pairs(arow0 + k0, half);
        v16bf a1 = load_frag_pairs(arow1 + k0, half);
        v16bf bf[4];
#pragma unroll
        for (int t = 0; t < 4; ++t)
            bf[t] = load_frag_pairs(brow + (size_t)(16 * t) * D_N + k0, half);
#pragma unroll
        for (int t = 0; t < 4; ++t) {
            c[0][t] = wmma_bf16(a0, bf[t], c[0][t]);
            c[1][t] = wmma_bf16(a1, bf[t], c[1][t]);
        }
        // pipeline hint: 12 vmem reads / 8 wmma per iter -> (3 loads, 2 wmma) x4
#pragma unroll
        for (int sgb = 0; sgb < 4; ++sgb) {
            SCHED_GROUP(0x020, 3, 0);
            SCHED_GROUP(0x008, 2, 0);
        }
    }

#pragma unroll
    for (int t = 0; t < 4; ++t) {
        float bval = bias[n0 + 16 * t + idx];
#pragma unroll
        for (int ms = 0; ms < 2; ++ms) {
#pragma unroll
            for (int v = 0; v < 8; ++v) {
                float o = c[ms][t][v] + bval;
                int row = m0 + ms * 16 + v + 8 * half;   // global bs row
                int col = n0 + 16 * t + idx;
                unsigned short h = f2bf(o);
                if (which == 0)      qo[(size_t)row * D_N + col] = h;
                else if (which == 1) ko[(size_t)row * D_N + col] = h;
                else {
                    int bb = row >> 11, s = row & (S_N - 1);
                    vto[((size_t)(bb * D_N + col)) * S_N + s] = h;
                }
            }
        }
    }
}

// ------------------------------------------------------------ flash attention
// block = 4 waves, one 16-row query block per block. Wave w owns ctx D-slice [128w,128w+128).
__global__ __launch_bounds__(128) __attribute__((amdgpu_waves_per_eu(1)))
void flash_kernel(
        const unsigned short* __restrict__ qb, const unsigned short* __restrict__ kb,
        const unsigned short* __restrict__ vtb, const int* __restrict__ maskp,
        unsigned short* __restrict__ ctxb) {
    __shared__ float red[4][2][16][16];
    __shared__ unsigned short Pl[16][32];

    const int lane = threadIdx.x & 31;
    const int wave = threadIdx.x >> 5;
    const int idx = lane & 15, half = lane >> 4;
    const int bidx = blockIdx.x / (S_N / 16);
    const int q0 = (blockIdx.x % (S_N / 16)) * 16;
    const int d0 = wave * 128;
    const float scale = 0.044194173824159216f;   // 1/sqrt(512)
    const float NEG = -1e30f;

    // Q fragments for this wave's D-slice (K = d)
    v16bf qf[4];
    const unsigned short* qrow = qb + (size_t)(bidx * S_N + q0 + idx) * D_N + d0;
#pragma unroll
    for (int kk = 0; kk < 4; ++kk) qf[kk] = load_frag_pairs(qrow + kk * 32, half);

    v8f ctx[8] = {};
    float m_r[8], l_r[8];
#pragma unroll
    for (int v = 0; v < 8; ++v) { m_r[v] = NEG; l_r[v] = 0.0f; }

    const int nkb = (q0 + 16 + 31) / 32;          // ceil((q0+16)/32) causal key blocks
    for (int kbi = 0; kbi < nkb; ++kbi) {
        const int j0 = kbi * 32;
        const unsigned short* krow0 = kb + (size_t)(bidx * S_N + j0 + idx) * D_N + d0;
        const unsigned short* krow1 = krow0 + (size_t)16 * D_N;

        // L2 warm-up for next key block (global_prefetch_b8)
        if (kbi + 1 < nkb) {
            __builtin_prefetch(krow0 + (size_t)32 * D_N, 0, 1);
            __builtin_prefetch(krow1 + (size_t)32 * D_N, 0, 1);
        }

        // issue ALL fragment loads for this key block up front: 8 K frags feed the
        // score WMMAs, 8 V frags are consumed only after the softmax, so they have
        // the whole LDS-reduction + softmax phase to land.
        v16bf kf0[4], kf1[4], vf[8];
#pragma unroll
        for (int kk = 0; kk < 4; ++kk) {
            kf0[kk] = load_frag_pairs(krow0 + kk * 32, half);
            kf1[kk] = load_frag_pairs(krow1 + kk * 32, half);
        }
#pragma unroll
        for (int t = 0; t < 8; ++t) {
            const unsigned short* vrow =
                vtb + (size_t)(bidx * D_N + d0 + 16 * t + idx) * S_N + j0;
            vf[t] = load_frag_pairs(vrow, half);
        }
        int mk0 = maskp[bidx * S_N + j0 + idx];
        int mk1 = maskp[bidx * S_N + j0 + 16 + idx];

        // partial scores over this wave's D-slice: S0 = cols j0..j0+15, S1 = +16..31
        v8f s0 = {}, s1 = {};
#pragma unroll
        for (int kk = 0; kk < 4; ++kk) {
            s0 = wmma_bf16(qf[kk], kf0[kk], s0);
            s1 = wmma_bf16(qf[kk], kf1[kk], s1);
        }
        // pipeline hint: 32+ vmem reads / 8 wmma in this region -> (4 loads, 1 wmma) x8
#pragma unroll
        for (int sgb = 0; sgb < 8; ++sgb) {
            SCHED_GROUP(0x020, 4, 0);
            SCHED_GROUP(0x008, 1, 0);
        }

#pragma unroll
        for (int v = 0; v < 8; ++v) {
            int r = v + 8 * half;
            red[wave][0][r][idx] = s0[v];
            red[wave][1][r][idx] = s1[v];
        }
        __syncthreads();

        // every wave reconstructs the full 16x32 score tile (identical math)
        float f0[8], f1[8];
#pragma unroll
        for (int v = 0; v < 8; ++v) {
            int r = v + 8 * half;
            float a0 = red[0][0][r][idx] + red[1][0][r][idx] + red[2][0][r][idx] + red[3][0][r][idx];
            float a1 = red[0][1][r][idx] + red[1][1][r][idx] + red[2][1][r][idx] + red[3][1][r][idx];
            int row = q0 + r;
            f0[v] = (mk0 == 0 || (j0 + idx)      > row) ? NEG : a0 * scale;
            f1[v] = (mk1 == 0 || (j0 + 16 + idx) > row) ? NEG : a1 * scale;
        }
        // online softmax per row (rows live in 16-lane groups -> xor shuffles stay in-group)
#pragma unroll
        for (int v = 0; v < 8; ++v) {
            float rm = fmaxf(f0[v], f1[v]);
#pragma unroll
            for (int sh = 1; sh < 16; sh <<= 1) rm = fmaxf(rm, __shfl_xor(rm, sh, 32));
            float mn = fmaxf(m_r[v], rm);
            float p0 = __expf(f0[v] - mn);
            float p1 = __expf(f1[v] - mn);
            float rs = p0 + p1;
#pragma unroll
            for (int sh = 1; sh < 16; sh <<= 1) rs += __shfl_xor(rs, sh, 32);
            float corr = __expf(m_r[v] - mn);
            l_r[v] = l_r[v] * corr + rs;
            m_r[v] = mn;
            f0[v] = p0; f1[v] = p1;
#pragma unroll
            for (int t = 0; t < 8; ++t) ctx[t][v] *= corr;
        }
        // wave 0 publishes P (bf16) so everyone can read it in A-fragment layout
        if (wave == 0) {
#pragma unroll
            for (int v = 0; v < 8; ++v) {
                int r = v + 8 * half;
                Pl[r][idx]      = f2bf(f0[v]);
                Pl[r][16 + idx] = f2bf(f1[v]);
            }
        }
        __syncthreads();

        v16bf pf = load_frag_pairs(&Pl[idx][0], half);
#pragma unroll
        for (int t = 0; t < 8; ++t) ctx[t] = wmma_bf16(pf, vf[t], ctx[t]);
    }

#pragma unroll
    for (int v = 0; v < 8; ++v) l_r[v] = 1.0f / l_r[v];
#pragma unroll
    for (int t = 0; t < 8; ++t) {
#pragma unroll
        for (int v = 0; v < 8; ++v) {
            int row = q0 + v + 8 * half;
            int col = d0 + 16 * t + idx;
            ctxb[(size_t)(bidx * S_N + row) * D_N + col] = f2bf(ctx[t][v] * l_r[v]);
        }
    }
}

// ---------------------------------------------------------- output projection
// grid BS/32, block 256 (8 waves). 32x64 tile per wave, fp32 out.
__global__ __launch_bounds__(256) __attribute__((amdgpu_waves_per_eu(1)))
void outproj_kernel(
        const unsigned short* __restrict__ ctxb, const unsigned short* __restrict__ wob,
        float* __restrict__ out) {
    const int lane = threadIdx.x & 31;
    const int wave = threadIdx.x >> 5;
    const int idx = lane & 15, half = lane >> 4;
    const int m0 = blockIdx.x * 32;
    const int n0 = wave * 64;

    v8f c[2][4] = {};
    const unsigned short* arow0 = ctxb + (size_t)(m0 + idx) * D_N;
    const unsigned short* arow1 = arow0 + (size_t)16 * D_N;
    const unsigned short* brow  = wob + (size_t)(n0 + idx) * D_N;

#pragma unroll
    for (int k0 = 0; k0 < D_N; k0 += 32) {
        v16bf a0 = load_frag_pairs(arow0 + k0, half);
        v16bf a1 = load_frag_pairs(arow1 + k0, half);
        v16bf bf[4];
#pragma unroll
        for (int t = 0; t < 4; ++t)
            bf[t] = load_frag_pairs(brow + (size_t)(16 * t) * D_N + k0, half);
#pragma unroll
        for (int t = 0; t < 4; ++t) {
            c[0][t] = wmma_bf16(a0, bf[t], c[0][t]);
            c[1][t] = wmma_bf16(a1, bf[t], c[1][t]);
        }
#pragma unroll
        for (int sgb = 0; sgb < 4; ++sgb) {
            SCHED_GROUP(0x020, 3, 0);
            SCHED_GROUP(0x008, 2, 0);
        }
    }

#pragma unroll
    for (int t = 0; t < 4; ++t)
#pragma unroll
        for (int ms = 0; ms < 2; ++ms)
#pragma unroll
            for (int v = 0; v < 8; ++v)
                out[(size_t)(m0 + ms * 16 + v + 8 * half) * D_N + n0 + 16 * t + idx] = c[ms][t][v];
}

extern "C" void kernel_launch(void* const* d_in, const int* in_sizes, int n_in,
                              void* d_out, int out_size, void* d_ws, size_t ws_size,
                              hipStream_t stream) {
    const float* x  = (const float*)d_in[0];
    const float* Wq = (const float*)d_in[1];
    const float* bq = (const float*)d_in[2];
    const float* Wk = (const float*)d_in[3];
    const float* bk = (const float*)d_in[4];
    const float* Wv = (const float*)d_in[5];
    const float* bv = (const float*)d_in[6];
    const float* Wo = (const float*)d_in[7];
    const int*  msk = (const int*)d_in[8];
    float* out = (float*)d_out;

    const size_t NX = (size_t)BS_N * D_N;   // 4,194,304 elems
    const size_t NW = (size_t)D_N * D_N;    //   262,144 elems
    unsigned short* xb   = (unsigned short*)d_ws;
    unsigned short* wqb  = xb  + NX;
    unsigned short* wkb  = wqb + NW;
    unsigned short* wvb  = wkb + NW;
    unsigned short* wob  = wvb + NW;
    unsigned short* qb   = wob + NW;
    unsigned short* kbuf = qb  + NX;
    unsigned short* vtb  = kbuf + NX;
    unsigned short* ctxb = vtb + NX;
    // total workspace: (5*NX + 4*NW)*2 bytes ~= 42 MB

    cvt_kernel<<<(int)((NX / 4 + 255) / 256), 256, 0, stream>>>(x,  xb,  (int)(NX / 4));
    cvt_kernel<<<(int)((NW / 4 + 255) / 256), 256, 0, stream>>>(Wq, wqb, (int)(NW / 4));
    cvt_kernel<<<(int)((NW / 4 + 255) / 256), 256, 0, stream>>>(Wk, wkb, (int)(NW / 4));
    cvt_kernel<<<(int)((NW / 4 + 255) / 256), 256, 0, stream>>>(Wv, wvb, (int)(NW / 4));
    cvt_kernel<<<(int)((NW / 4 + 255) / 256), 256, 0, stream>>>(Wo, wob, (int)(NW / 4));

    qkv_kernel<<<dim3(BS_N / 32, 3), 256, 0, stream>>>(xb, wqb, wkb, wvb, bq, bk, bv,
                                                       qb, kbuf, vtb);
    flash_kernel<<<BS_N / 16, 128, 0, stream>>>(qb, kbuf, vtb, msk, ctxb);
    outproj_kernel<<<BS_N / 32, 256, 0, stream>>>(ctxb, wob, out);
}
